// ConvAttnPoolPlusGram_79044578116371
// MI455X (gfx1250) — matrix operand order repaired and checked by reading, not compiled, not run on hardware
//
#include <hip/hip_runtime.h>
#include <hip/hip_bf16.h>

// ---- problem dims ----
#define B_   8
#define N_   2500
#define E_   100
#define K_   60
#define NP_  6
#define Y_   8921
#define F_   50
#define KS_  10
#define H_   20
#define PADc 5
#define L_   2501
// padded dims for WMMA tiling
#define LP_  2528      // multiple of 32, >= 2512 (157*16)
#define YP_  8928      // 558*16
#define FP_  64        // padded F
#define NYT_ 558
#define NLT_ 157
#define NK2_ 79        // LP_/32
#define NYG_ 70        // ceil(NYT_/8)

typedef _Float16 v16h __attribute__((ext_vector_type(16)));
typedef _Float16 h8   __attribute__((ext_vector_type(8)));
typedef float    v8f  __attribute__((ext_vector_type(8)));
// TDM descriptor groups (clang-23 amdgpu-toolchain 6-arg builtin form)
typedef unsigned int u32x4 __attribute__((ext_vector_type(4)));
typedef int          i32x8 __attribute__((ext_vector_type(8)));
typedef int          i32x4 __attribute__((ext_vector_type(4)));

// ---------------- kernel 1: word embedding gather ----------------
__global__ void k_embed(const int* __restrict__ x, const float* __restrict__ embed_W,
                        float* __restrict__ xe) {
    int idx = blockIdx.x * blockDim.x + threadIdx.x;
    if (idx >= B_ * N_ * E_) return;
    int bn = idx / E_, e = idx - bn * E_;
    int tok = x[bn];
    xe[idx] = embed_W[(long long)tok * E_ + e];
}

// ---------------- kernel 2: concept attention + recombine + replace ----------------
__global__ void k_concept(const int* __restrict__ concepts, const int* __restrict__ parents,
                          const int* __restrict__ dm,
                          const float* __restrict__ embed_W, const float* __restrict__ concept_W,
                          const float* __restrict__ fc1_W, const float* __restrict__ fc1_b,
                          const float* __restrict__ fc2_W, const float* __restrict__ fc2_b,
                          const float* __restrict__ rec_W, const float* __restrict__ rec_b,
                          float* __restrict__ xe) {
    __shared__ float sp[NP_ * E_];      // parent embeddings
    __shared__ float sh1[NP_ * H_];
    __shared__ float salpha[NP_];
    __shared__ float sc[E_];
    __shared__ float sdme[E_];
    int b = blockIdx.x / K_, k = blockIdx.x % K_;
    int tid = threadIdx.x;
    for (int idx = tid; idx < NP_ * E_; idx += blockDim.x) {
        int j = idx / E_, e = idx - j * E_;
        int pid = parents[(b * K_ + k) * NP_ + j];
        sp[idx] = concept_W[(long long)pid * E_ + e];
    }
    for (int e = tid; e < E_; e += blockDim.x)
        sdme[e] = embed_W[(long long)dm[b * K_ + k] * E_ + e];
    __syncthreads();
    if (tid < NP_ * H_) {
        int j = tid / H_, h = tid - j * H_;
        float acc = fc1_b[h];
        const float* w = fc1_W + h * (2 * E_);
        for (int e = 0; e < E_; ++e) acc += sp[e] * w[e];               // child = parent 0
        for (int e = 0; e < E_; ++e) acc += sp[j * E_ + e] * w[E_ + e];
        sh1[tid] = tanhf(acc);
    }
    __syncthreads();
    if (tid < NP_) {
        float s = fc2_b[0];
        for (int h = 0; h < H_; ++h) s += sh1[tid * H_ + h] * fc2_W[h];
        salpha[tid] = s;
    }
    __syncthreads();
    if (tid == 0) {
        float m = salpha[0];
        for (int j = 1; j < NP_; ++j) m = fmaxf(m, salpha[j]);
        float s = 0.f;
        for (int j = 0; j < NP_; ++j) { salpha[j] = __expf(salpha[j] - m); s += salpha[j]; }
        float inv = 1.f / s;
        for (int j = 0; j < NP_; ++j) salpha[j] *= inv;
    }
    __syncthreads();
    for (int e = tid; e < E_; e += blockDim.x) {
        float cv = 0.f;
        for (int j = 0; j < NP_; ++j) cv += salpha[j] * sp[j * E_ + e];
        sc[e] = cv;
    }
    __syncthreads();
    if (concepts[b * N_ + k] > 0) {
        for (int e = tid; e < E_; e += blockDim.x) {
            float acc = rec_b[e];
            const float* w = rec_W + e * (2 * E_);
            for (int i = 0; i < E_; ++i) acc += sc[i] * w[i] + sdme[i] * w[E_ + i];
            xe[((long long)b * N_ + k) * E_ + e] = acc;
        }
    }
}

// ---------------- kernel 3: conv1d + tanh -> padded f16 h [B][LP_][FP_] ----------------
__global__ void k_conv(const float* __restrict__ xe, const float* __restrict__ conv_W,
                       const float* __restrict__ conv_b, _Float16* __restrict__ hf) {
    __shared__ float win[KS_ * E_];
    int b = blockIdx.x / LP_, l = blockIdx.x % LP_;
    int f = threadIdx.x;
    for (int idx = threadIdx.x; idx < KS_ * E_; idx += blockDim.x) {
        int r = idx / E_, e = idx - r * E_;
        int src = l - PADc + r;
        win[idx] = (l < L_ && src >= 0 && src < N_) ? xe[((long long)b * N_ + src) * E_ + e] : 0.f;
    }
    __syncthreads();
    _Float16 out = (_Float16)0.f;
    if (l < L_ && f < F_) {
        float acc = conv_b[f];
        const float* w = conv_W + (long long)f * E_ * KS_;
        for (int e = 0; e < E_; ++e) {
            #pragma unroll
            for (int k = 0; k < KS_; ++k) acc += win[k * E_ + e] * w[e * KS_ + k];
        }
        out = (_Float16)tanhf(acc);
    }
    hf[((long long)b * LP_ + l) * FP_ + f] = out;
}

// ---------------- kernel 4: U_W -> padded f16 [YP_][FP_] ----------------
__global__ void k_u16(const float* __restrict__ U_W, _Float16* __restrict__ Uf) {
    int idx = blockIdx.x * blockDim.x + threadIdx.x;
    if (idx >= YP_ * FP_) return;
    int y = idx / FP_, f = idx - y * FP_;
    Uf[idx] = (y < Y_ && f < F_) ? (_Float16)U_W[(long long)y * F_ + f] : (_Float16)0.f;
}

__global__ void k_zero(float* p) { *p = 0.f; }

// ---------------- kernel 5: GEMM1 scores[b,y,l] = sum_f U[y,f]*h[b,l,f] (WMMA) ----------------
__global__ __launch_bounds__(256) void k_scores(const _Float16* __restrict__ Uf,
                                                const _Float16* __restrict__ hf,
                                                float* __restrict__ att) {
    int wid = blockIdx.x * 8 + (threadIdx.x >> 5);
    if (wid >= B_ * NYT_ * NLT_) return;
    int lane = threadIdx.x & 31;
    int lt = wid % NLT_;
    int yt = (wid / NLT_) % NYT_;
    int b  = wid / (NLT_ * NYT_);
    int half = lane >> 4, lm = lane & 15;

    const _Float16* Urow = Uf + (long long)(yt * 16 + lm) * FP_;
    const _Float16* Hrow = hf + ((long long)b * LP_ + (lt * 16 + lm)) * FP_;

    v8f c = {0.f, 0.f, 0.f, 0.f, 0.f, 0.f, 0.f, 0.f};
    #pragma unroll
    for (int s = 0; s < 2; ++s) {
        int fbase = s * 32;
        // A fragment (16x32 f16): elems 0..7 = K fbase+8*half.., elems 8..15 = K fbase+16+8*half..
        h8 alo = *(const h8*)(Urow + fbase + 8 * half);
        h8 ahi = *(const h8*)(Urow + fbase + 16 + 8 * half);
        v16h a;
        #pragma unroll
        for (int i = 0; i < 8; ++i) { a[i] = alo[i]; a[8 + i] = ahi[i]; }
        // B fragment (32x16 f16): elems 0..7 = K fbase+8*half.., elems 8..15 = K fbase+16+8*half.., col = lm
        h8 blo = *(const h8*)(Hrow + fbase + 8 * half);
        h8 bhi = *(const h8*)(Hrow + fbase + 16 + 8 * half);
        v16h bb;
        #pragma unroll
        for (int i = 0; i < 8; ++i) { bb[i] = blo[i]; bb[8 + i] = bhi[i]; }
        c = __builtin_amdgcn_wmma_f32_16x16x32_f16(false, a, false, bb, (short)0, c, false, false);
    }
    int lcol = lt * 16 + lm;
    float* orow = att + ((long long)b * Y_ + yt * 16 + 8 * half) * L_ + lcol;
    if (yt < 557 && lt < 156) {
        // fully interior tile: unguarded coalesced stores (covers ~99% of tiles)
        #pragma unroll
        for (int r = 0; r < 8; ++r) orow[(long long)r * L_] = c[r];
    } else {
        #pragma unroll
        for (int r = 0; r < 8; ++r) {
            int y = yt * 16 + r + 8 * half;
            if (y < Y_ && lcol < L_) orow[(long long)r * L_] = c[r];
        }
    }
}

// ---------------- kernel 6: row softmax over L, in place ----------------
__global__ __launch_bounds__(256) void k_softmax(float* __restrict__ att) {
    __shared__ float red[256];
    float* row = att + (long long)blockIdx.x * L_;
    int tid = threadIdx.x;
    float vals[10];
    float m = -3.402823e38f;
    #pragma unroll
    for (int j = 0; j < 10; ++j) {
        int i = tid + 256 * j;
        if (i < L_) { vals[j] = row[i]; m = fmaxf(m, vals[j]); }
    }
    red[tid] = m; __syncthreads();
    for (int s = 128; s > 0; s >>= 1) {
        if (tid < s) red[tid] = fmaxf(red[tid], red[tid + s]);
        __syncthreads();
    }
    m = red[0]; __syncthreads();
    float ssum = 0.f;
    #pragma unroll
    for (int j = 0; j < 10; ++j) {
        int i = tid + 256 * j;
        if (i < L_) { vals[j] = __expf(vals[j] - m); ssum += vals[j]; }
    }
    red[tid] = ssum; __syncthreads();
    for (int s = 128; s > 0; s >>= 1) {
        if (tid < s) red[tid] += red[tid + s];
        __syncthreads();
    }
    float inv = 1.f / red[0];
    #pragma unroll
    for (int j = 0; j < 10; ++j) {
        int i = tid + 256 * j;
        if (i < L_) row[i] = vals[j] * inv;
    }
}

// ---------------- kernel 7: GEMM2 m = att @ h (WMMA, TDM-staged LDS) fused with final head ----------------
__global__ __launch_bounds__(256) void k_final(const float* __restrict__ att,
                                               const _Float16* __restrict__ hf,
                                               const float* __restrict__ final_W,
                                               const float* __restrict__ final_b,
                                               const float* __restrict__ target,
                                               float* __restrict__ yhat,
                                               float* __restrict__ lossAcc) {
    __shared__ __align__(16) _Float16 sB[2][32 * FP_];   // double-buffered TDM landing zone
    int b  = blockIdx.x / NYG_;
    int yg = blockIdx.x % NYG_;
    int wid = threadIdx.x >> 5, lane = threadIdx.x & 31;
    int half = lane >> 4, lm = lane & 15;
    int yt = yg * 8 + wid;
    bool wvalid = (yt < NYT_);
    if (!wvalid) yt = NYT_ - 1;
    bool issuer = (threadIdx.x < 32);   // wave 0 drives the Tensor Data Mover

    const float* attb = att + (long long)b * Y_ * L_;
    int yA = yt * 16 + lm;
    if (yA >= Y_) yA = Y_ - 1;
    const float* arow = attb + (long long)yA * L_;

    // Issue a TDM descriptor: DMA h[b][kbase..kbase+31][0..63] (32x64 f16 tile) -> LDS buffer.
    auto tdm_issue = [&](int ksNext, int buf) {
        unsigned ldsOff = (unsigned)(unsigned long long)&sB[buf][0];
        unsigned long long ga =
            (unsigned long long)(const void*)(hf + ((long long)b * LP_ + ksNext * 32) * FP_);
        u32x4 g0;
        g0[0] = 1u;                                   // count=1, user mode, no gather
        g0[1] = ldsOff;                               // lds_addr (bytes)
        g0[2] = (unsigned)(ga & 0xFFFFFFFFu);         // global_addr[31:0]
        g0[3] = (unsigned)((ga >> 32) & 0x1FFFFFFu)   // global_addr[56:32]
              | (2u << 30);                           // type=2 ("image")
        i32x8 g1;
        g1[0] = 0x10000;                              // workgroup_mask=0, data_size=1 (2 bytes)
        g1[1] = (int)(64u << 16);                     // tensor_dim0=64 (low 16 at [31:16])
        g1[2] = (int)(((unsigned)LP_ & 0xFFFFu) << 16); // td0 hi=0 | tensor_dim1=LP_ low16
        g1[3] = (int)(64u << 16);                     // td1 hi=0 | tile_dim0=64
        g1[4] = 32;                                   // tile_dim1=32, tile_dim2=0
        g1[5] = 64;                                   // tensor_dim0_stride=64 (low 32)
        g1[6] = 0;                                    // t0s hi16=0 | t1s low16=0
        g1[7] = 0;                                    // t1s hi=0
        i32x4 z4 = {0, 0, 0, 0};                      // groups 2/3 unused (2D tile)
        i32x8 z8 = {0, 0, 0, 0, 0, 0, 0, 0};          // extra group (6-arg toolchain form)
        __builtin_amdgcn_tensor_load_to_lds(g0, g1, z4, z4, z8, 0);
    };

    v8f acc[4];
    #pragma unroll
    for (int t = 0; t < 4; ++t) acc[t] = (v8f){0.f,0.f,0.f,0.f,0.f,0.f,0.f,0.f};

    if (issuer) tdm_issue(0, 0);
    for (int ks = 0; ks < NK2_; ++ks) {
        int kbase = ks * 32;
        if (issuer) __builtin_amdgcn_s_wait_tensorcnt(0);   // tile ks landed in LDS
        __syncthreads();                                    // publish to all waves
        if (issuer && ks + 1 < NK2_)
            tdm_issue(ks + 1, (ks + 1) & 1);                // DMA next tile during compute
        const _Float16* sb = &sB[ks & 1][0];

        // A fragment from att row (f32 -> f16)
        v16h a;
        #pragma unroll
        for (int i = 0; i < 8; ++i) {
            int l0 = kbase + 8 * half + i;      if (l0 > L_ - 1) l0 = L_ - 1;
            int l1 = kbase + 16 + 8 * half + i; if (l1 > L_ - 1) l1 = L_ - 1;
            a[i]     = (_Float16)arow[l0];
            a[8 + i] = (_Float16)arow[l1];
        }
        // 4 B fragments from LDS (column reads per the 32x16 B layout) + WMMA
        #pragma unroll
        for (int t4 = 0; t4 < 4; ++t4) {
            v16h bb;
            #pragma unroll
            for (int i = 0; i < 16; ++i)
                bb[i] = sb[(8 * half + (i & 7) + ((i >> 3) << 4)) * FP_ + 16 * t4 + lm];
            acc[t4] = __builtin_amdgcn_wmma_f32_16x16x32_f16(false, a, false, bb, (short)0,
                                                             acc[t4], false, false);
        }
        if (ks + 1 < NK2_) __builtin_prefetch(arow + kbase + 32, 0, 1);
        // next iteration's barrier protects buffer reuse
    }

    if (wvalid) {
        int col = lm;
        #pragma unroll
        for (int r = 0; r < 8; ++r) {
            int y = yt * 16 + r + 8 * half;   // uniform across each 16-lane half
            float partial = 0.f;
            if (y < Y_) {
                const float* fw = final_W + (long long)y * F_;
                #pragma unroll
                for (int t4 = 0; t4 < 4; ++t4) {
                    int f = t4 * 16 + col;
                    if (f < F_) partial += acc[t4][r] * fw[f];
                }
            }
            partial += __shfl_xor(partial, 1, 32);
            partial += __shfl_xor(partial, 2, 32);
            partial += __shfl_xor(partial, 4, 32);
            partial += __shfl_xor(partial, 8, 32);
            if (col == 0 && y < Y_) {
                float yv = partial + final_b[y];
                float yh = 1.f / (1.f + __expf(-yv));
                yhat[(long long)b * Y_ + y] = yh;
                float pc = fminf(fmaxf(yh, 1e-7f), 1.f - 1e-7f);
                float tg = target[(long long)b * Y_ + y];
                atomicAdd(lossAcc, tg * __logf(pc) + (1.f - tg) * log1pf(-pc));
            }
        }
    }
}

__global__ void k_loss_finish(const float* lossAcc, float* lossOut) {
    *lossOut = -(*lossAcc) / (float)(B_ * Y_);
}

// ---------------- launcher ----------------
extern "C" void kernel_launch(void* const* d_in, const int* in_sizes, int n_in,
                              void* d_out, int out_size, void* d_ws, size_t ws_size,
                              hipStream_t stream) {
    const int*   x         = (const int*)d_in[0];
    const int*   concepts  = (const int*)d_in[1];
    const int*   parents   = (const int*)d_in[2];
    // d_in[3] = batched_concepts_mask (unused by the reference math)
    const int*   dm        = (const int*)d_in[4];
    const float* target    = (const float*)d_in[5];
    const float* embed_W   = (const float*)d_in[6];
    const float* concept_W = (const float*)d_in[7];
    const float* fc1_W     = (const float*)d_in[8];
    const float* fc1_b     = (const float*)d_in[9];
    const float* fc2_W     = (const float*)d_in[10];
    const float* fc2_b     = (const float*)d_in[11];
    const float* rec_W     = (const float*)d_in[12];
    const float* rec_b     = (const float*)d_in[13];
    const float* conv_W    = (const float*)d_in[14];
    const float* conv_b    = (const float*)d_in[15];
    const float* U_W       = (const float*)d_in[16];
    const float* final_W   = (const float*)d_in[17];
    const float* final_b   = (const float*)d_in[18];

    char* ws = (char*)d_ws;
    float*    xe      = (float*)ws;                          // 8,000,000 B
    _Float16* hf      = (_Float16*)(ws + 8000000);           // B*LP*FP*2 = 2,588,672 B
    _Float16* Uf      = (_Float16*)(ws + 10588672);          // YP*FP*2   = 1,142,784 B
    float*    lossAcc = (float*)(ws + 11731456);

    float* yhat    = (float*)d_out;
    float* lossOut = yhat + (long long)B_ * Y_;
    float* att     = lossOut + 1;

    k_embed<<<(B_ * N_ * E_ + 255) / 256, 256, 0, stream>>>(x, embed_W, xe);
    k_concept<<<B_ * K_, 128, 0, stream>>>(concepts, parents, dm, embed_W, concept_W,
                                           fc1_W, fc1_b, fc2_W, fc2_b, rec_W, rec_b, xe);
    k_conv<<<B_ * LP_, 64, 0, stream>>>(xe, conv_W, conv_b, hf);
    k_u16<<<(YP_ * FP_ + 255) / 256, 256, 0, stream>>>(U_W, Uf);
    k_zero<<<1, 1, 0, stream>>>(lossAcc);
    k_scores<<<(B_ * NYT_ * NLT_ + 7) / 8, 256, 0, stream>>>(Uf, hf, att);
    k_softmax<<<B_ * Y_, 256, 0, stream>>>(att);
    k_final<<<B_ * NYG_, 256, 0, stream>>>(att, hf, final_W, final_b, target, yhat, lossAcc);
    k_loss_finish<<<1, 1, 0, stream>>>(lossAcc, lossOut);
}
